// DuplexAttn_59820304499405
// MI455X (gfx1250) — matrix operand order repaired and checked by reading, not compiled, not run on hardware
//
#include <hip/hip_runtime.h>
#include <math.h>

typedef __bf16 bf16;
typedef __attribute__((ext_vector_type(16))) __bf16 v16bf;
typedef __attribute__((ext_vector_type(8)))  float  v8f;
typedef int v4i __attribute__((vector_size(16)));

#if defined(__HIP_DEVICE_COMPILE__) && __has_builtin(__builtin_amdgcn_global_load_async_to_lds_b128)
#define HAVE_ASYNC_LDS 1
#else
#define HAVE_ASYNC_LDS 0
#endif

// ---------------------------------------------------------------- helpers
__device__ inline v8f zero8() {
  v8f z;
#pragma unroll
  for (int i = 0; i < 8; ++i) z[i] = 0.f;
  return z;
}

// 16-byte per-lane copy global -> LDS (async on CDNA5, sync fallback otherwise)
__device__ inline void cp16_g2l(bf16* lds_dst, const bf16* gsrc) {
#if HAVE_ASYNC_LDS
  __builtin_amdgcn_global_load_async_to_lds_b128(
      (__attribute__((address_space(1))) v4i*)gsrc,
      (__attribute__((address_space(3))) v4i*)lds_dst, 0, 0);
#else
  *(uint4*)lds_dst = *(const uint4*)gsrc;
#endif
}
__device__ inline void async_wait_all() {
#if HAVE_ASYNC_LDS
#if __has_builtin(__builtin_amdgcn_s_wait_asynccnt)
  __builtin_amdgcn_s_wait_asynccnt(0);
#else
  asm volatile("s_wait_asynccnt 0" ::: "memory");
#endif
#endif
}

// A fragment, 16x32 bf16, row-major source (lda in elements).
// Lane l(0..15)=row M=l holds K in {kb..kb+7, 16+kb..16+kb+7}, kb=8*(l>>4).
__device__ inline v16bf frag_a_rm(const bf16* __restrict__ A, int lda, int lane) {
  int m = lane & 15, kb = (lane >> 4) * 8;
  const bf16* p = A + (size_t)m * lda + kb;
  v16bf a;
#pragma unroll
  for (int r = 0; r < 4; ++r) { a[2*r] = p[2*r]; a[2*r+1] = p[2*r+1]; }
#pragma unroll
  for (int r = 0; r < 4; ++r) { a[8+2*r] = p[16+2*r]; a[8+2*r+1] = p[16+2*r+1]; }
  return a;
}

// A fragment from a 32-column LDS tile (lda = 32)
__device__ inline v16bf frag_a_lds(const bf16* S, int lane) {
  int m = lane & 15, kb = (lane >> 4) * 8;
  const bf16* p = S + m * 32 + kb;
  v16bf a;
#pragma unroll
  for (int r = 0; r < 4; ++r) { a[2*r] = p[2*r]; a[2*r+1] = p[2*r+1]; }
#pragma unroll
  for (int r = 0; r < 4; ++r) { a[8+2*r] = p[16+2*r]; a[8+2*r+1] = p[16+2*r+1]; }
  return a;
}

// A fragment where logical A(m,k) = src[k*lds + m]  (i.e. A = srcᵀ tile)
__device__ inline v16bf frag_a_cm(const bf16* __restrict__ S, int lds, int lane) {
  int m = lane & 15, kb = (lane >> 4) * 8;
  const bf16* p = S + (size_t)kb * lds + m;
  v16bf a;
#pragma unroll
  for (int r = 0; r < 4; ++r) { a[2*r] = p[(size_t)(2*r)*lds]; a[2*r+1] = p[(size_t)(2*r+1)*lds]; }
#pragma unroll
  for (int r = 0; r < 4; ++r) { a[8+2*r] = p[(size_t)(16+2*r)*lds]; a[8+2*r+1] = p[(size_t)(16+2*r+1)*lds]; }
  return a;
}

// B fragment (B = Wᵀ) from row-major W[N,K]: lane n holds B column n = W row n.
__device__ inline v16bf frag_b_nk(const bf16* __restrict__ W, int ldw, int lane) {
  int n = lane & 15, kb = (lane >> 4) * 8;
  const bf16* p = W + (size_t)n * ldw + kb;
  v16bf b;
#pragma unroll
  for (int j = 0; j < 16; ++j) b[j] = p[(j & 7) + ((j >> 3) << 4)];
  return b;
}

// B fragment from row-major B[K,N] itself (column loads, stride ldb).
__device__ inline v16bf frag_b_kn(const bf16* __restrict__ Bm, int ldb, int lane) {
  int n = lane & 15, kb = (lane >> 4) * 8;
  const bf16* p = Bm + (size_t)kb * ldb + n;
  v16bf b;
#pragma unroll
  for (int j = 0; j < 16; ++j) b[j] = p[(size_t)((j & 7) + ((j >> 3) << 4)) * ldb];
  return b;
}

__device__ inline v8f wmma_bf16(v16bf a, v16bf b, v8f c) {
  return __builtin_amdgcn_wmma_f32_16x16x32_bf16(false, a, false, b, (short)0, c, false, false);
}

// ---------------------------------------------------------------- LN(cond)
__global__ void k_ln_cond(const float* __restrict__ cond, const float* __restrict__ g,
                          const float* __restrict__ bb, float* __restrict__ of,
                          bf16* __restrict__ ob) {
  int row = blockIdx.x;            // 4096
  int t = threadIdx.x;             // 256 threads, 2 elems each
  const float* src = cond + (size_t)row * 512;
  float v0 = src[t], v1 = src[t + 256];
  __shared__ float red[256];
  red[t] = v0 + v1; __syncthreads();
  for (int s = 128; s > 0; s >>= 1) { if (t < s) red[t] += red[t + s]; __syncthreads(); }
  float mean = red[0] * (1.f / 512.f); __syncthreads();
  float d0 = v0 - mean, d1 = v1 - mean;
  red[t] = d0 * d0 + d1 * d1; __syncthreads();
  for (int s = 128; s > 0; s >>= 1) { if (t < s) red[t] += red[t + s]; __syncthreads(); }
  float rstd = rsqrtf(red[0] * (1.f / 512.f) + 1e-5f);
  float o0 = d0 * rstd * g[t] + bb[t];
  float o1 = d1 * rstd * g[t + 256] + bb[t + 256];
  size_t o = (size_t)row * 512;
  of[o + t] = o0; of[o + t + 256] = o1;
  ob[o + t] = (bf16)o0; ob[o + t + 256] = (bf16)o1;
}

// ---------------------------------------------------------------- converts
__global__ void k_cvt(const float* __restrict__ s, bf16* __restrict__ d, int n) {
  int i = blockIdx.x * blockDim.x + threadIdx.x;
  if (i < n) d[i] = (bf16)s[i];
}

__global__ void k_cvt_x(const float* __restrict__ x, bf16* __restrict__ xt) {
  size_t i = (size_t)blockIdx.x * blockDim.x + threadIdx.x;
  if (i >= (size_t)16 * 256 * 1024) return;
  int p = (int)(i & 1023);
  size_t r = i >> 10;
  int c = (int)(r & 255), b = (int)(r >> 8);
  xt[((size_t)b * 1024 + p) * 256 + c] = (bf16)x[i];
}

__global__ void k_cvt_wout(const float* __restrict__ w, bf16* __restrict__ d) {
  int i = blockIdx.x * blockDim.x + threadIdx.x;   // 512*512*9
  if (i >= 512 * 512 * 9) return;
  int kx = i % 3; int t = i / 3; int ky = t % 3; t /= 3;
  int ic = t % 512; int oc = t / 512;
  d[(size_t)oc * 4608 + (ky * 3 + kx) * 512 + ic] = (bf16)w[i];
}

// ------------------------------------------------ WMMA GEMM (C = A·Wᵀ + bias)
// block = 128 thr (4 waves); block tile 32x256; wave tile 32x64 (8 wmma/K-step)
// A 32x32 chunk staged in LDS (async global->LDS), shared by all 4 waves.
__global__ void __launch_bounds__(128)
k_gemm(const bf16* __restrict__ A, int lda, const bf16* __restrict__ W, int ldw,
       const float* __restrict__ bias, float* __restrict__ C, int ldc, int K) {
  __shared__ __align__(16) bf16 As[32 * 32];
  int lane = threadIdx.x & 31, wave = threadIdx.x >> 5;
  int n0 = blockIdx.x * 256 + wave * 64;
  int m0 = blockIdx.y * 32;
  int crow = threadIdx.x >> 2, cseg = (threadIdx.x & 3) * 8;   // 16B per thread
  const bf16* Asrc = A + (size_t)(m0 + crow) * lda + cseg;
  v8f acc[2][4];
#pragma unroll
  for (int i = 0; i < 2; ++i)
#pragma unroll
    for (int j = 0; j < 4; ++j) acc[i][j] = zero8();
  for (int k = 0; k < K; k += 32) {
    cp16_g2l(&As[crow * 32 + cseg], Asrc + k);
    async_wait_all();
    __syncthreads();
    v16bf a0 = frag_a_lds(As, lane);
    v16bf a1 = frag_a_lds(As + 16 * 32, lane);
#pragma unroll
    for (int j = 0; j < 4; ++j) {
      const bf16* wp = W + (size_t)(n0 + j * 16) * ldw + k;
      __builtin_prefetch(wp + 32, 0, 3);
      v16bf b = frag_b_nk(wp, ldw, lane);
      acc[0][j] = wmma_bf16(a0, b, acc[0][j]);
      acc[1][j] = wmma_bf16(a1, b, acc[1][j]);
    }
    __syncthreads();
  }
  int n = lane & 15, hi = lane >> 4;
#pragma unroll
  for (int j = 0; j < 4; ++j) {
    float bv = bias ? bias[n0 + j * 16 + n] : 0.f;
#pragma unroll
    for (int i = 0; i < 2; ++i)
#pragma unroll
      for (int r = 0; r < 8; ++r)
        C[(size_t)(m0 + i * 16 + hi * 8 + r) * ldc + n0 + j * 16 + n] = acc[i][j][r] + bv;
  }
}

// -------------------------------------- l2norm + fold exp(min(scale,ln100)) into q
__global__ void k_norm_q(const float* __restrict__ qv, const float* __restrict__ ss,
                         bf16* __restrict__ qn, bf16* __restrict__ vq) {
  int i = blockIdx.x * blockDim.x + threadIdx.x;  // 16384*8
  if (i >= 16384 * 8) return;
  int h = i & 7, t = i >> 3, b = t >> 10, s = t & 1023;
  const float* q = qv + (size_t)t * 1024 + h * 64;
  const float* v = q + 512;
  float sum = 0.f;
  for (int d = 0; d < 64; ++d) { float x = q[d]; sum += x * x; }
  float inv = 1.f / fmaxf(sqrtf(sum), 1e-12f);
  float scale = __expf(fminf(ss[h], 4.6051702f));   // ln(100)
  size_t o = ((size_t)(b * 8 + h) * 1024 + s) * 64;
  for (int d = 0; d < 64; ++d) { qn[o + d] = (bf16)(q[d] * inv * scale); vq[o + d] = (bf16)v[d]; }
}

__global__ void k_norm_k(const float* __restrict__ kv, bf16* __restrict__ kn,
                         bf16* __restrict__ vk) {
  int i = blockIdx.x * blockDim.x + threadIdx.x;  // 4096*8
  if (i >= 4096 * 8) return;
  int h = i & 7, t = i >> 3, b = t >> 8, j = t & 255;
  const float* k = kv + (size_t)t * 1024 + h * 64;
  const float* v = k + 512;
  float sum = 0.f;
  for (int d = 0; d < 64; ++d) { float x = k[d]; sum += x * x; }
  float inv = 1.f / fmaxf(sqrtf(sum), 1e-12f);
  size_t o = ((size_t)(b * 8 + h) * 256 + j) * 64;
  for (int d = 0; d < 64; ++d) { kn[o + d] = (bf16)(k[d] * inv); vk[o + d] = (bf16)v[d]; }
}

// --------------------- logits (16x256) via WMMA + in-register softmax -> bf16 probs
__global__ void __launch_bounds__(32)
k_attn(const bf16* __restrict__ qn, const bf16* __restrict__ kn, bf16* __restrict__ attn) {
  int lane = threadIdx.x;
  int bh = blockIdx.y;                // 128
  int m0 = blockIdx.x * 16;           // over 1024
  const bf16* Q = qn + ((size_t)bh * 1024 + m0) * 64;
  const bf16* Kb = kn + (size_t)bh * 256 * 64;
  v8f acc[16];
#pragma unroll
  for (int t = 0; t < 16; ++t) acc[t] = zero8();
  for (int kk = 0; kk < 64; kk += 32) {
    v16bf a = frag_a_rm(Q + kk, 64, lane);
#pragma unroll
    for (int t = 0; t < 16; ++t) {
      v16bf b = frag_b_nk(Kb + (size_t)t * 16 * 64 + kk, 64, lane);
      acc[t] = wmma_bf16(a, b, acc[t]);
    }
  }
  int n = lane & 15, hi = lane >> 4;
#pragma unroll
  for (int r = 0; r < 8; ++r) {
    float mx = -3.4e38f;
#pragma unroll
    for (int t = 0; t < 16; ++t) mx = fmaxf(mx, acc[t][r]);
    for (int d = 8; d >= 1; d >>= 1) mx = fmaxf(mx, __shfl_xor(mx, d, 16));
    float sum = 0.f;
#pragma unroll
    for (int t = 0; t < 16; ++t) { float e = __expf(acc[t][r] - mx); acc[t][r] = e; sum += e; }
    for (int d = 8; d >= 1; d >>= 1) sum += __shfl_xor(sum, d, 16);
    float is = 1.f / sum;
    size_t row = (size_t)bh * 1024 + m0 + hi * 8 + r;
#pragma unroll
    for (int t = 0; t < 16; ++t) attn[row * 256 + t * 16 + n] = (bf16)(acc[t][r] * is);
  }
}

// ---------------------------------------- qkv = attn · v_k  -> merged tokens bf16
__global__ void __launch_bounds__(32)
k_av(const bf16* __restrict__ attn, const bf16* __restrict__ vk, bf16* __restrict__ qkv) {
  int lane = threadIdx.x;
  int bh = blockIdx.y, m0 = blockIdx.x * 16;
  int b = bh >> 3, h = bh & 7;
  const bf16* A = attn + ((size_t)bh * 1024 + m0) * 256;
  const bf16* V = vk + (size_t)bh * 256 * 64;
  v8f acc[4];
#pragma unroll
  for (int t = 0; t < 4; ++t) acc[t] = zero8();
  for (int kk = 0; kk < 256; kk += 32) {
    v16bf a = frag_a_rm(A + kk, 256, lane);
#pragma unroll
    for (int t = 0; t < 4; ++t) {
      v16bf bf = frag_b_kn(V + (size_t)kk * 64 + t * 16, 64, lane);
      acc[t] = wmma_bf16(a, bf, acc[t]);
    }
  }
  int n = lane & 15, hi = lane >> 4;
#pragma unroll
  for (int t = 0; t < 4; ++t)
#pragma unroll
    for (int r = 0; r < 8; ++r) {
      int s = m0 + hi * 8 + r;
      qkv[((size_t)b * 1024 + s) * 512 + h * 64 + t * 16 + n] = (bf16)acc[t][r];
    }
}

// ---------------------------------------- vkq = attnᵀ · v_q -> merged tokens bf16
__global__ void __launch_bounds__(32)
k_atv(const bf16* __restrict__ attn, const bf16* __restrict__ vq, bf16* __restrict__ vkq) {
  int lane = threadIdx.x;
  int bh = blockIdx.y, j0 = blockIdx.x * 16;   // over s_c=256
  int b = bh >> 3, h = bh & 7;
  const bf16* A = attn + (size_t)bh * 1024 * 256 + j0; // A(m,k)=attn[k*256 + j0+m]
  const bf16* V = vq + (size_t)bh * 1024 * 64;
  v8f acc[4];
#pragma unroll
  for (int t = 0; t < 4; ++t) acc[t] = zero8();
  for (int kk = 0; kk < 1024; kk += 32) {
    v16bf a = frag_a_cm(A + (size_t)kk * 256, 256, lane);
#pragma unroll
    for (int t = 0; t < 4; ++t) {
      v16bf bf = frag_b_kn(V + (size_t)kk * 64 + t * 16, 64, lane);
      acc[t] = wmma_bf16(a, bf, acc[t]);
    }
  }
  int n = lane & 15, hi = lane >> 4;
#pragma unroll
  for (int t = 0; t < 4; ++t)
#pragma unroll
    for (int r = 0; r < 8; ++r) {
      int j = j0 + hi * 8 + r;
      vkq[((size_t)b * 256 + j) * 512 + h * 64 + t * 16 + n] = (bf16)acc[t][r];
    }
}

// ------------------------- 3x3 SAME conv as implicit-GEMM (K = 9*512) via WMMA
// block tile 32x256 (tokens x oc); wave tile 32x64; A chunk staged in LDS
// (async) with boundary zero-fill; m0 tile = one full image row (y const).
__global__ void __launch_bounds__(128)
k_conv(const bf16* __restrict__ qkv, const bf16* __restrict__ w, float* __restrict__ out) {
  __shared__ __align__(16) bf16 As[32 * 32];
  int lane = threadIdx.x & 31, wave = threadIdx.x >> 5;
  int n0 = blockIdx.x * 256 + wave * 64;     // oc
  int m0 = blockIdx.y * 32;                  // token rows, 16384/32 blocks
  int b = m0 >> 10, y = (m0 & 1023) >> 5;
  int crow = threadIdx.x >> 2, cseg = (threadIdx.x & 3) * 8;  // crow = x (= m in tile)
  v8f acc[2][4];
#pragma unroll
  for (int i = 0; i < 2; ++i)
#pragma unroll
    for (int j = 0; j < 4; ++j) acc[i][j] = zero8();
  for (int kp = 0; kp < 9; ++kp) {
    int ky = kp / 3 - 1, kx = kp % 3 - 1;
    int ny = y + ky, nx = crow + kx;
    bool valid = (ny >= 0) && (ny < 32) && (nx >= 0) && (nx < 32);
    const bf16* Asrc = valid ? qkv + ((size_t)b * 1024 + ny * 32 + nx) * 512 + cseg : nullptr;
    for (int kc = 0; kc < 512; kc += 32) {
      if (valid) {
        cp16_g2l(&As[crow * 32 + cseg], Asrc + kc);
      } else {
        uint4 z = {0u, 0u, 0u, 0u};
        *(uint4*)&As[crow * 32 + cseg] = z;
      }
      async_wait_all();
      __syncthreads();
      v16bf a0 = frag_a_lds(As, lane);
      v16bf a1 = frag_a_lds(As + 16 * 32, lane);
#pragma unroll
      for (int j = 0; j < 4; ++j) {
        const bf16* wp = w + (size_t)(n0 + j * 16) * 4608 + kp * 512 + kc;
        __builtin_prefetch(wp + 32, 0, 3);
        v16bf bw = frag_b_nk(wp, 4608, lane);
        acc[0][j] = wmma_bf16(a0, bw, acc[0][j]);
        acc[1][j] = wmma_bf16(a1, bw, acc[1][j]);
      }
      __syncthreads();
    }
  }
  int n = lane & 15, hi = lane >> 4;
#pragma unroll
  for (int j = 0; j < 4; ++j)
#pragma unroll
    for (int i = 0; i < 2; ++i)
#pragma unroll
      for (int r = 0; r < 8; ++r)
        out[(size_t)(m0 + i * 16 + hi * 8 + r) * 512 + n0 + j * 16 + n] = acc[i][j][r];
}

// ------------- channel-LN over 512, split scale/shift, out = shift + x*scale
__global__ void k_ln_out(const float* __restrict__ conv, const float* __restrict__ g,
                         const float* __restrict__ bb, const float* __restrict__ x,
                         float* __restrict__ out) {
  int t = blockIdx.x;        // 16384 tokens
  int c = threadIdx.x;       // 256
  const float* src = conv + (size_t)t * 512;
  float v0 = src[c], v1 = src[c + 256];
  __shared__ float red[256];
  red[c] = v0 + v1; __syncthreads();
  for (int s = 128; s > 0; s >>= 1) { if (c < s) red[c] += red[c + s]; __syncthreads(); }
  float mean = red[0] * (1.f / 512.f); __syncthreads();
  float d0 = v0 - mean, d1 = v1 - mean;
  red[c] = d0 * d0 + d1 * d1; __syncthreads();
  for (int s = 128; s > 0; s >>= 1) { if (c < s) red[c] += red[c + s]; __syncthreads(); }
  float rstd = rsqrtf(red[0] * (1.f / 512.f) + 1e-5f);
  float sc = d0 * rstd * g[c] + bb[c];
  float sh = d1 * rstd * g[c + 256] + bb[c + 256];
  int b = t >> 10, p = t & 1023;
  size_t xi = ((size_t)b * 256 + c) * 1024 + p;
  out[xi] = sh + x[xi] * sc;
}

// ------------- LN(cu)*g+b + cond_n  -> cond_out
__global__ void k_ln_cu(const float* __restrict__ cu, const float* __restrict__ g,
                        const float* __restrict__ bb, const float* __restrict__ condn,
                        float* __restrict__ out) {
  int row = blockIdx.x;      // 4096
  int c = threadIdx.x;       // 256
  const float* src = cu + (size_t)row * 512;
  float v0 = src[c], v1 = src[c + 256];
  __shared__ float red[256];
  red[c] = v0 + v1; __syncthreads();
  for (int s = 128; s > 0; s >>= 1) { if (c < s) red[c] += red[c + s]; __syncthreads(); }
  float mean = red[0] * (1.f / 512.f); __syncthreads();
  float d0 = v0 - mean, d1 = v1 - mean;
  red[c] = d0 * d0 + d1 * d1; __syncthreads();
  for (int s = 128; s > 0; s >>= 1) { if (c < s) red[c] += red[c + s]; __syncthreads(); }
  float rstd = rsqrtf(red[0] * (1.f / 512.f) + 1e-5f);
  size_t o = (size_t)row * 512;
  out[o + c]       = condn[o + c]       + d0 * rstd * g[c] + bb[c];
  out[o + c + 256] = condn[o + c + 256] + d1 * rstd * g[c + 256] + bb[c + 256];
}

// ---------------------------------------------------------------- launch
extern "C" void kernel_launch(void* const* d_in, const int* in_sizes, int n_in,
                              void* d_out, int out_size, void* d_ws, size_t ws_size,
                              hipStream_t stream) {
  (void)in_sizes; (void)n_in; (void)out_size; (void)ws_size;
  const float* x    = (const float*)d_in[0];
  const float* cond = (const float*)d_in[1];
  const float* w_qv = (const float*)d_in[2];
  const float* b_qv = (const float*)d_in[3];
  const float* w_kv = (const float*)d_in[4];
  const float* b_kv = (const float*)d_in[5];
  const float* ss   = (const float*)d_in[6];
  const float* w_out= (const float*)d_in[7];
  const float* g_out= (const float*)d_in[8];
  const float* b_out= (const float*)d_in[9];
  const float* w_cu = (const float*)d_in[10];
  const float* g_cu = (const float*)d_in[11];
  const float* b_cu = (const float*)d_in[12];
  const float* g_cn = (const float*)d_in[13];
  const float* b_cn = (const float*)d_in[14];
  float* out = (float*)d_out;                       // [16,256,32,32] then [16,256,512]
  float* cond_out = out + (size_t)16 * 256 * 1024;

  char* w = (char*)d_ws;
  auto take = [&](size_t bytes) { char* p = w; w += (bytes + 255) & ~(size_t)255; return p; };
  bf16* x_tok  = (bf16*)take((size_t)16384 * 256 * 2);
  bf16* wqvb   = (bf16*)take((size_t)1024 * 256 * 2);
  bf16* wkvb   = (bf16*)take((size_t)1024 * 512 * 2);
  bf16* wcub   = (bf16*)take((size_t)512 * 512 * 2);
  bf16* woutb  = (bf16*)take((size_t)512 * 4608 * 2);
  float* condn_f = (float*)take((size_t)4096 * 512 * 4);
  bf16* condn_b  = (bf16*)take((size_t)4096 * 512 * 2);
  float* qv_f  = (float*)take((size_t)16384 * 1024 * 4);
  float* kv_f  = (float*)take((size_t)4096 * 1024 * 4);
  bf16* qn     = (bf16*)take((size_t)16 * 8 * 1024 * 64 * 2);
  bf16* kn     = (bf16*)take((size_t)16 * 8 * 256 * 64 * 2);
  bf16* vq     = (bf16*)take((size_t)16 * 8 * 1024 * 64 * 2);
  bf16* vk     = (bf16*)take((size_t)16 * 8 * 256 * 64 * 2);
  bf16* attn   = (bf16*)take((size_t)16 * 8 * 1024 * 256 * 2);
  bf16* qkvtok = (bf16*)take((size_t)16384 * 512 * 2);
  bf16* vkqtok = (bf16*)take((size_t)4096 * 512 * 2);
  float* conv_f = (float*)take((size_t)16384 * 512 * 4);
  float* cu_f   = (float*)take((size_t)4096 * 512 * 4);

  // 1. LN(cond)
  k_ln_cond<<<4096, 256, 0, stream>>>(cond, g_cn, b_cn, condn_f, condn_b);
  // 2. converts
  k_cvt_x<<<(16 * 256 * 1024 + 255) / 256, 256, 0, stream>>>(x, x_tok);
  k_cvt<<<(1024 * 256 + 255) / 256, 256, 0, stream>>>(w_qv, wqvb, 1024 * 256);
  k_cvt<<<(1024 * 512 + 255) / 256, 256, 0, stream>>>(w_kv, wkvb, 1024 * 512);
  k_cvt<<<(512 * 512 + 255) / 256, 256, 0, stream>>>(w_cu, wcub, 512 * 512);
  k_cvt_wout<<<(512 * 512 * 9 + 255) / 256, 256, 0, stream>>>(w_out, woutb);
  // 3. qv = x_tok · Wqvᵀ + b   [16384,1024], K=256
  k_gemm<<<dim3(1024 / 256, 16384 / 32), 128, 0, stream>>>(x_tok, 256, wqvb, 256, b_qv,
                                                           qv_f, 1024, 256);
  // 4. kv = cond_n · Wkvᵀ + b  [4096,1024], K=512
  k_gemm<<<dim3(1024 / 256, 4096 / 32), 128, 0, stream>>>(condn_b, 512, wkvb, 512, b_kv,
                                                          kv_f, 1024, 512);
  // 5. l2norm + scale fold
  k_norm_q<<<(16384 * 8 + 255) / 256, 256, 0, stream>>>(qv_f, ss, qn, vq);
  k_norm_k<<<(4096 * 8 + 255) / 256, 256, 0, stream>>>(kv_f, kn, vk);
  // 6. attention probs
  k_attn<<<dim3(64, 128), 32, 0, stream>>>(qn, kn, attn);
  // 7. qkv = attn·v_k, vkq = attnᵀ·v_q
  k_av<<<dim3(64, 128), 32, 0, stream>>>(attn, vk, qkvtok);
  k_atv<<<dim3(16, 128), 32, 0, stream>>>(attn, vq, vkqtok);
  // 8. 3x3 conv (implicit GEMM, K=4608)
  k_conv<<<dim3(512 / 256, 16384 / 32), 128, 0, stream>>>(qkvtok, woutb, conv_f);
  // 9. channel LN + addcmul -> out
  k_ln_out<<<16384, 256, 0, stream>>>(conv_f, g_out, b_out, x, out);
  // 10. cu = vkq · Wcuᵀ  [4096,512], K=512
  k_gemm<<<dim3(512 / 256, 4096 / 32), 128, 0, stream>>>(vkqtok, 512, wcub, 512, nullptr,
                                                         cu_f, 512, 512);
  // 11. LN(cu) + cond_n -> cond_out
  k_ln_cu<<<4096, 256, 0, stream>>>(cu_f, g_cu, b_cu, condn_f, cond_out);
}
